// MixtureOfExperts_72816875536958
// MI455X (gfx1250) — matrix operand order repaired and checked by reading, not compiled, not run on hardware
//
#include <hip/hip_runtime.h>

// ---------------------------------------------------------------------------
// MoE (E=8, top-2) for MI455X / gfx1250.
// Heavy math: v_wmma_f32_16x16x32_bf16 (wave32 WMMA).
// Operand movement: global_load_async_to_lds_b128 + s_wait_asynccnt,
// double-buffered LDS slabs feeding contiguous ds_load_b128 fragments.
// ---------------------------------------------------------------------------

typedef __attribute__((ext_vector_type(16))) __bf16 v16bf;
typedef __attribute__((ext_vector_type(8)))  __bf16 v8bf;
typedef __attribute__((ext_vector_type(8)))  float  v8f;
typedef __attribute__((ext_vector_type(4)))  int    v4i;
typedef __bf16 bf16;

#define AS_GLOBAL __attribute__((address_space(1)))
#define AS_LDS    __attribute__((address_space(3)))

union AFrag { v16bf v; v8bf h[2]; };

static constexpr int NTOK  = 4096;   // B*S
static constexpr int HD    = 1024;   // hidden
static constexpr int ID    = 2048;   // intermediate
static constexpr int NE    = 8;      // experts
static constexpr int NSLOT = NTOK * 2;

__device__ __forceinline__ v8f wmma_bf16(v16bf a, v16bf b, v8f c) {
  return __builtin_amdgcn_wmma_f32_16x16x32_bf16(false, a, false, b,
                                                 (short)0, c, false, false);
}

// --- async copy global(16B) -> LDS, tracked by ASYNCcnt ---------------------
#if __has_builtin(__builtin_amdgcn_global_load_async_to_lds_b128)
#define ASYNC_CP(dstLds, srcGlb)                                              \
  __builtin_amdgcn_global_load_async_to_lds_b128(                             \
      (AS_GLOBAL v4i*)(srcGlb), (AS_LDS v4i*)(dstLds), 0, 0)
#else
#define ASYNC_CP(dstLds, srcGlb)                                              \
  asm volatile("global_load_async_to_lds_b128 %0, %1, off"                    \
               :: "v"((unsigned)(unsigned long long)(dstLds)),                \
                  "v"((unsigned long long)(srcGlb)) : "memory")
#endif

#if __has_builtin(__builtin_amdgcn_s_wait_asynccnt)
#define WAIT_ASYNC(n) __builtin_amdgcn_s_wait_asynccnt(n)
#else
#define WAIT_ASYNC(n) asm volatile("s_wait_asynccnt %0" :: "i"(n) : "memory")
#endif

// LDS fragment loads (contiguous per-lane 32B / 16B)
__device__ __forceinline__ v16bf lds_load16(const bf16* p) {
  return *(const AS_LDS v16bf*)(p);
}
__device__ __forceinline__ v8bf lds_load8(const bf16* p) {
  return *(const AS_LDS v8bf*)(p);
}

// ---------------------------------------------------------------------------
// Router: one wave32 per token.
// ---------------------------------------------------------------------------
__global__ __launch_bounds__(256) void moe_router(
    const float* __restrict__ x, const float* __restrict__ wr,
    bf16* __restrict__ xbf, int* __restrict__ sel, float* __restrict__ rw,
    int* __restrict__ counts)
{
  const int wave = threadIdx.x >> 5;
  const int lane = threadIdx.x & 31;
  const int n = blockIdx.x * 8 + wave;
  const float* xr = x + (size_t)n * HD;

  float acc[NE];
#pragma unroll
  for (int e = 0; e < NE; ++e) acc[e] = 0.f;

  for (int hh = lane; hh < HD; hh += 32) {
    float xv = xr[hh];
    xbf[(size_t)n * HD + hh] = (bf16)xv;
#pragma unroll
    for (int e = 0; e < NE; ++e) acc[e] += xv * wr[e * HD + hh];
  }
#pragma unroll
  for (int off = 16; off > 0; off >>= 1) {
#pragma unroll
    for (int e = 0; e < NE; ++e) acc[e] += __shfl_down(acc[e], off, 32);
  }
  if (lane == 0) {
    float mx = acc[0];
#pragma unroll
    for (int e = 1; e < NE; ++e) mx = fmaxf(mx, acc[e]);
    float p[NE];
#pragma unroll
    for (int e = 0; e < NE; ++e) p[e] = __expf(acc[e] - mx);
    int i0 = 0;
#pragma unroll
    for (int e = 1; e < NE; ++e) if (p[e] > p[i0]) i0 = e;
    int i1 = (i0 == 0) ? 1 : 0;
#pragma unroll
    for (int e = 0; e < NE; ++e) if (e != i0 && p[e] > p[i1]) i1 = e;
    const float s2 = p[i0] + p[i1];
    sel[2 * n + 0] = i0;  rw[2 * n + 0] = p[i0] / s2;
    sel[2 * n + 1] = i1;  rw[2 * n + 1] = p[i1] / s2;
    atomicAdd(&counts[i0], 1);
    atomicAdd(&counts[i1], 1);
  }
}

__global__ void moe_zero_counts(int* counts) {
  if (threadIdx.x < NE) counts[threadIdx.x] = 0;
}

__global__ void moe_scan(const int* __restrict__ counts, int* __restrict__ offsets,
                         int* __restrict__ cursor) {
  if (threadIdx.x == 0) {
    int run = 0;
    for (int e = 0; e < NE; ++e) { offsets[e] = run; run += counts[e]; cursor[e] = 0; }
  }
}

__global__ __launch_bounds__(256) void moe_assign(
    const int* __restrict__ sel, const float* __restrict__ rw,
    int* __restrict__ cursor, const int* __restrict__ offsets,
    int* __restrict__ tok_of_slot, float* __restrict__ w_of_slot,
    int* __restrict__ slots)
{
  const int n = blockIdx.x * 256 + threadIdx.x;
  if (n >= NTOK) return;
#pragma unroll
  for (int k = 0; k < 2; ++k) {
    const int e = sel[2 * n + k];
    const int pos = atomicAdd(&cursor[e], 1);
    const int slot = offsets[e] + pos;
    tok_of_slot[slot] = n;
    w_of_slot[slot] = rw[2 * n + k];
    slots[2 * n + k] = slot;
  }
}

// Deterministic aux loss: aux = 0.01 * sum_e S_e^2 / N (fixed-order tree).
__global__ __launch_bounds__(256) void moe_aux(
    const int* __restrict__ sel, const float* __restrict__ rw, float* __restrict__ out)
{
  __shared__ float red[NE][256];
  float a[NE];
#pragma unroll
  for (int e = 0; e < NE; ++e) a[e] = 0.f;
  for (int i = threadIdx.x; i < NSLOT; i += 256) a[sel[i]] += rw[i];
#pragma unroll
  for (int e = 0; e < NE; ++e) red[e][threadIdx.x] = a[e];
  __syncthreads();
  for (int s = 128; s > 0; s >>= 1) {
    if ((int)threadIdx.x < s) {
#pragma unroll
      for (int e = 0; e < NE; ++e) red[e][threadIdx.x] += red[e][threadIdx.x + s];
    }
    __syncthreads();
  }
  if (threadIdx.x == 0) {
    float t = 0.f;
#pragma unroll
    for (int e = 0; e < NE; ++e) { float S = red[e][0]; t += S * S; }
    out[(size_t)NTOK * HD] = 0.01f * t / (float)NTOK;
  }
}

// ---------------------------------------------------------------------------
// Transpose+convert: (E, R, C) f32 -> (E, C, R) bf16.
// ---------------------------------------------------------------------------
__global__ __launch_bounds__(256) void transpose_convert(
    const float* __restrict__ in, bf16* __restrict__ out, int R, int C)
{
  __shared__ float tile[32][33];
  const size_t mat = (size_t)R * C;
  const float* src = in + (size_t)blockIdx.z * mat;
  bf16* dst = out + (size_t)blockIdx.z * mat;
  const int c = blockIdx.x * 32 + threadIdx.x;
#pragma unroll
  for (int i = 0; i < 4; ++i) {
    const int r = blockIdx.y * 32 + threadIdx.y + i * 8;
    tile[threadIdx.y + i * 8][threadIdx.x] = src[(size_t)r * C + c];
  }
  __syncthreads();
  const int r2 = blockIdx.y * 32 + threadIdx.x;
#pragma unroll
  for (int i = 0; i < 4; ++i) {
    const int c2 = blockIdx.x * 32 + threadIdx.y + i * 8;
    dst[(size_t)c2 * R + r2] = (bf16)tile[threadIdx.x][threadIdx.y + i * 8];
  }
}

// ---------------------------------------------------------------------------
// GEMM1: per-expert X_e @ [Wg | Wu], fused SiLU(g)*u -> h (bf16).
// Block tile 128M x 64N, 8 waves = (wm 0..3) x (wn 0..1); each wave 32x32
// per matrix (8 accumulators total). K staged in 64-wide double-buffered
// LDS slabs via async-to-LDS copies.
// ---------------------------------------------------------------------------
__global__ __launch_bounds__(256) void moe_gate_up(
    const bf16* __restrict__ xbf, const bf16* __restrict__ wgt,
    const bf16* __restrict__ wut, const float* __restrict__ bg,
    const float* __restrict__ bu, const int* __restrict__ tok_of_slot,
    const int* __restrict__ offsets, const int* __restrict__ counts,
    bf16* __restrict__ hbuf)
{
  __shared__ bf16 shA[2][128 * 64];       // [buf][row*64 + k]
  __shared__ bf16 shB[2][2][64 * 64];     // [buf][mat][col*64 + k]

  const int e = blockIdx.z;
  const int ne = counts[e];
  const int m0 = (int)blockIdx.y * 128;
  if (m0 >= ne) return;
  const int base = offsets[e];
  const int rowEnd = base + ne;
  const int c0 = blockIdx.x * 64;
  const int tid = threadIdx.x;
  const int wave = tid >> 5, lane = tid & 31;
  const int wm = wave & 3, wn = wave >> 2;
  const int laneLo = lane & 15, laneHi = lane >> 4;

  // Precompute gathered token per A-chunk (row fixed per chunk).
  int aTok[4];
#pragma unroll
  for (int i = 0; i < 4; ++i) {
    const int chunk = tid + 256 * i;          // 0..1023 (128 rows x 8 chunks)
    const int row = chunk >> 3;
    int srow = base + m0 + row;
    srow = (srow < rowEnd) ? srow : rowEnd - 1;
    aTok[i] = tok_of_slot[srow];
  }

  auto stage = [&](int k0, int buf) {
#pragma unroll
    for (int i = 0; i < 4; ++i) {             // A: 128x64 bf16 = 16KB
      const int chunk = tid + 256 * i;
      const int koff = (chunk & 7) * 8;
      const bf16* g = xbf + (size_t)aTok[i] * HD + k0 + koff;
      ASYNC_CP(&shA[buf][chunk * 8], g);
    }
#pragma unroll
    for (int i = 0; i < 2; ++i) {             // B: 2 x 64x64 bf16 = 16KB
      const int chunk = tid + 256 * i;        // 0..511
      const int col = chunk >> 3;
      const int koff = (chunk & 7) * 8;
      const size_t rowOff = ((size_t)e * ID + c0 + col) * HD + k0 + koff;
      ASYNC_CP(&shB[buf][0][chunk * 8], wgt + rowOff);
      ASYNC_CP(&shB[buf][1][chunk * 8], wut + rowOff);
    }
  };

  v8f acc[2][2][2];                           // [mat][mt][nt]
#pragma unroll
  for (int a = 0; a < 2; ++a)
#pragma unroll
    for (int b = 0; b < 2; ++b)
#pragma unroll
      for (int c = 0; c < 2; ++c) acc[a][b][c] = {};

  stage(0, 0);
  for (int kk = 0; kk < HD; kk += 64) {
    const int cur = (kk >> 6) & 1;
    if (kk + 64 < HD) { stage(kk + 64, cur ^ 1); WAIT_ASYNC(8); }
    else              { WAIT_ASYNC(0); }
    __syncthreads();
#pragma unroll
    for (int ks = 0; ks < 2; ++ks) {
      const int kb = ks * 32;
      AFrag af[2];
#pragma unroll
      for (int mt = 0; mt < 2; ++mt) {
        const bf16* ap = &shA[cur][(wm * 32 + mt * 16 + laneLo) * 64 + kb + laneHi * 16];
        af[mt].h[0] = lds_load8(ap);
        af[mt].h[1] = lds_load8(ap + 8);
      }
      v16bf bf_[2][2];
#pragma unroll
      for (int mat = 0; mat < 2; ++mat)
#pragma unroll
        for (int nt = 0; nt < 2; ++nt)
          bf_[mat][nt] = lds_load16(
              &shB[cur][mat][(wn * 32 + nt * 16 + laneLo) * 64 + kb + laneHi * 16]);
#pragma unroll
      for (int mat = 0; mat < 2; ++mat)
#pragma unroll
        for (int mt = 0; mt < 2; ++mt)
#pragma unroll
          for (int nt = 0; nt < 2; ++nt)
            acc[mat][mt][nt] = wmma_bf16(af[mt].v, bf_[mat][nt], acc[mat][mt][nt]);
    }
    __syncthreads();
  }

#pragma unroll
  for (int mt = 0; mt < 2; ++mt) {
#pragma unroll
    for (int nt = 0; nt < 2; ++nt) {
      const int col = c0 + wn * 32 + nt * 16 + laneLo;
      const float bgv = bg[e * ID + col];
      const float buv = bu[e * ID + col];
#pragma unroll
      for (int r = 0; r < 8; ++r) {
        const int srow = base + m0 + wm * 32 + mt * 16 + r + 8 * laneHi;
        if (srow < rowEnd) {
          float gv = acc[0][mt][nt][r] + bgv;
          gv = gv / (1.f + __expf(-gv));       // SiLU
          const float uv = acc[1][mt][nt][r] + buv;
          hbuf[(size_t)srow * ID + col] = (bf16)(gv * uv);
        }
      }
    }
  }
}

// ---------------------------------------------------------------------------
// GEMM2: per-expert (g*u) @ Wd; routing weight + bias folded into epilogue.
// Block tile 128M x 64N, same async-LDS pipeline (6 copies/thread/slab).
// ---------------------------------------------------------------------------
__global__ __launch_bounds__(256) void moe_down(
    const bf16* __restrict__ hbuf, const bf16* __restrict__ wdt,
    const float* __restrict__ bd, const float* __restrict__ w_of_slot,
    const int* __restrict__ offsets, const int* __restrict__ counts,
    float* __restrict__ ybuf)
{
  __shared__ bf16 shA[2][128 * 64];
  __shared__ bf16 shB[2][64 * 64];

  const int e = blockIdx.z;
  const int ne = counts[e];
  const int m0 = (int)blockIdx.y * 128;
  if (m0 >= ne) return;
  const int base = offsets[e];
  const int rowEnd = base + ne;
  const int c0 = blockIdx.x * 64;
  const int tid = threadIdx.x;
  const int wave = tid >> 5, lane = tid & 31;
  const int wm = wave & 3, wn = wave >> 2;
  const int laneLo = lane & 15, laneHi = lane >> 4;

  int aRowC[4];
#pragma unroll
  for (int i = 0; i < 4; ++i) {
    const int chunk = tid + 256 * i;
    const int row = chunk >> 3;
    int srow = base + m0 + row;
    aRowC[i] = (srow < rowEnd) ? srow : rowEnd - 1;
  }

  auto stage = [&](int k0, int buf) {
#pragma unroll
    for (int i = 0; i < 4; ++i) {
      const int chunk = tid + 256 * i;
      const int koff = (chunk & 7) * 8;
      const bf16* g = hbuf + (size_t)aRowC[i] * ID + k0 + koff;
      ASYNC_CP(&shA[buf][chunk * 8], g);
    }
#pragma unroll
    for (int i = 0; i < 2; ++i) {
      const int chunk = tid + 256 * i;
      const int col = chunk >> 3;
      const int koff = (chunk & 7) * 8;
      const bf16* g = wdt + ((size_t)e * HD + c0 + col) * ID + k0 + koff;
      ASYNC_CP(&shB[buf][chunk * 8], g);
    }
  };

  v8f acc[2][2];                             // [mt][nt]
#pragma unroll
  for (int a = 0; a < 2; ++a)
#pragma unroll
    for (int b = 0; b < 2; ++b) acc[a][b] = {};

  stage(0, 0);
  for (int kk = 0; kk < ID; kk += 64) {
    const int cur = (kk >> 6) & 1;
    if (kk + 64 < ID) { stage(kk + 64, cur ^ 1); WAIT_ASYNC(6); }
    else              { WAIT_ASYNC(0); }
    __syncthreads();
#pragma unroll
    for (int ks = 0; ks < 2; ++ks) {
      const int kb = ks * 32;
      AFrag af[2];
#pragma unroll
      for (int mt = 0; mt < 2; ++mt) {
        const bf16* ap = &shA[cur][(wm * 32 + mt * 16 + laneLo) * 64 + kb + laneHi * 16];
        af[mt].h[0] = lds_load8(ap);
        af[mt].h[1] = lds_load8(ap + 8);
      }
      v16bf bf_[2];
#pragma unroll
      for (int nt = 0; nt < 2; ++nt)
        bf_[nt] = lds_load16(
            &shB[cur][(wn * 32 + nt * 16 + laneLo) * 64 + kb + laneHi * 16]);
#pragma unroll
      for (int mt = 0; mt < 2; ++mt)
#pragma unroll
        for (int nt = 0; nt < 2; ++nt)
          acc[mt][nt] = wmma_bf16(af[mt].v, bf_[nt], acc[mt][nt]);
    }
    __syncthreads();
  }

#pragma unroll
  for (int mt = 0; mt < 2; ++mt) {
#pragma unroll
    for (int nt = 0; nt < 2; ++nt) {
      const int col = c0 + wn * 32 + nt * 16 + laneLo;
      const float bdv = bd[e * HD + col];
#pragma unroll
      for (int r = 0; r < 8; ++r) {
        const int srow = base + m0 + wm * 32 + mt * 16 + r + 8 * laneHi;
        if (srow < rowEnd) {
          const float w = w_of_slot[srow];
          ybuf[(size_t)srow * HD + col] = w * (acc[mt][nt][r] + bdv);
        }
      }
    }
  }
}

// out[n,:] = y[slot0,:] + y[slot1,:]
__global__ __launch_bounds__(256) void moe_combine(
    const float* __restrict__ y, const int* __restrict__ slots,
    float* __restrict__ out)
{
  const size_t idx = (size_t)blockIdx.x * 256 + threadIdx.x;
  const int n = (int)(idx >> 10);           // HD == 1024
  const int hcol = (int)(idx & (HD - 1));
  const int s0 = slots[2 * n + 0];
  const int s1 = slots[2 * n + 1];
  out[idx] = y[(size_t)s0 * HD + hcol] + y[(size_t)s1 * HD + hcol];
}

// ---------------------------------------------------------------------------
extern "C" void kernel_launch(void* const* d_in, const int* in_sizes, int n_in,
                              void* d_out, int out_size, void* d_ws, size_t ws_size,
                              hipStream_t stream)
{
  (void)in_sizes; (void)n_in; (void)out_size; (void)ws_size;
  const float* x  = (const float*)d_in[0];
  const float* wr = (const float*)d_in[1];
  const float* wg = (const float*)d_in[2];
  const float* bg = (const float*)d_in[3];
  const float* wu = (const float*)d_in[4];
  const float* bu = (const float*)d_in[5];
  const float* wd = (const float*)d_in[6];
  const float* bd = (const float*)d_in[7];
  float* out = (float*)d_out;

  char* p = (char*)d_ws;
  auto take = [&](size_t bytes) {
    char* r = p;
    p += (bytes + 255) & ~(size_t)255;
    return r;
  };
  bf16*  xbf  = (bf16*)take((size_t)NTOK * HD * sizeof(bf16));
  bf16*  wgt  = (bf16*)take((size_t)NE * ID * HD * sizeof(bf16));
  bf16*  wut  = (bf16*)take((size_t)NE * ID * HD * sizeof(bf16));
  bf16*  wdt  = (bf16*)take((size_t)NE * HD * ID * sizeof(bf16));
  bf16*  hbuf = (bf16*)take((size_t)NSLOT * ID * sizeof(bf16));
  float* ybuf = (float*)take((size_t)NSLOT * HD * sizeof(float));
  int*   sel  = (int*)take(NSLOT * sizeof(int));
  float* rw   = (float*)take(NSLOT * sizeof(float));
  int*   slots = (int*)take(NSLOT * sizeof(int));
  int*   tok_of_slot = (int*)take(NSLOT * sizeof(int));
  float* w_of_slot   = (float*)take(NSLOT * sizeof(float));
  int*   counts  = (int*)take(NE * sizeof(int));
  int*   offsets = (int*)take(NE * sizeof(int));
  int*   cursor  = (int*)take(NE * sizeof(int));

  moe_zero_counts<<<dim3(1), dim3(32), 0, stream>>>(counts);
  moe_router<<<dim3(NTOK / 8), dim3(256), 0, stream>>>(x, wr, xbf, sel, rw, counts);
  transpose_convert<<<dim3(ID / 32, HD / 32, NE), dim3(32, 8), 0, stream>>>(wg, wgt, HD, ID);
  transpose_convert<<<dim3(ID / 32, HD / 32, NE), dim3(32, 8), 0, stream>>>(wu, wut, HD, ID);
  transpose_convert<<<dim3(HD / 32, ID / 32, NE), dim3(32, 8), 0, stream>>>(wd, wdt, ID, HD);
  moe_scan<<<dim3(1), dim3(1), 0, stream>>>(counts, offsets, cursor);
  moe_assign<<<dim3(NTOK / 256), dim3(256), 0, stream>>>(sel, rw, cursor, offsets,
                                                         tok_of_slot, w_of_slot, slots);
  moe_aux<<<dim3(1), dim3(256), 0, stream>>>(sel, rw, out);
  moe_gate_up<<<dim3(ID / 64, NTOK / 128, NE), dim3(256), 0, stream>>>(
      xbf, wgt, wut, bg, bu, tok_of_slot, offsets, counts, hbuf);
  moe_down<<<dim3(HD / 64, NTOK / 128, NE), dim3(256), 0, stream>>>(
      hbuf, wdt, bd, w_of_slot, offsets, counts, ybuf);
  moe_combine<<<dim3((NTOK * HD) / 256), dim3(256), 0, stream>>>(ybuf, slots, out);
}
